// HourglassTransformer_30399778521720
// MI455X (gfx1250) — matrix operand order repaired and checked by reading, not compiled, not run on hardware
//
#include <hip/hip_runtime.h>

typedef __attribute__((ext_vector_type(16))) _Float16 v16h;
typedef __attribute__((ext_vector_type(8)))  float    v8f;
typedef __attribute__((ext_vector_type(4)))  float    f32x4;

constexpr int  D     = 512;
constexpr int  DFF   = 2048;
constexpr int  NHEAD = 8;
constexpr int  HDIM  = 64;
constexpr int  SEQ   = 512;
constexpr int  NB    = 64;                 // batch
constexpr long SC_FLOATS = 16777216L;      // 64MB scratch: scores / FFN hidden

// ---------------------------------------------------------------------------
// WMMA GEMM:  C[M,N] = alpha * A[M,K] * (TRANSB ? Bm[N,K]^T : Bm[K,N]) + bias
//             (+ReLU) (+resid).  Batched via z: g = g0+z, b=g/HDIV, h=g%HDIV,
//             operand offset = b*s1 + h*s2 + z*s3 (all 32-bit element strides).
// Wave computes a 16x64 C tile (4 x v_wmma_f32_16x16x32_f16, f32 accum).
// GUARD paths use clamped-address loads + value select: branchless, no EXEC
// divergence around loads, so VMEM stays batched (no per-element waits).
// ---------------------------------------------------------------------------
template<bool TRANSB, bool RELU, bool GUARD>
__global__ __launch_bounds__(256) void gemm_k(
    const float* __restrict__ A, int lda, int sA1, int sA2, int sA3,
    const float* __restrict__ Bm, int ldb, int sB1, int sB2, int sB3,
    const float* __restrict__ bias,
    const float* __restrict__ Rp, int ldr, int sR1, int sR3,
    float* __restrict__ C, int ldc, int sC1, int sC2, int sC3,
    int M, int N, int Kd, float alpha, int g0, int HDIV)
{
  const int lane = threadIdx.x & 31;
  const int wave = threadIdx.x >> 5;
  const int m0 = (blockIdx.y * 8 + wave) * 16;
  if (m0 >= M) return;                       // wave-uniform: EXEC stays all-1s
  const int n0 = blockIdx.x * 64;
  const int z  = blockIdx.z;
  const int g  = g0 + z;
  const int bb = g / HDIV;
  const int hh = g - bb * HDIV;
  const float* Ab = A  + bb * sA1 + hh * sA2 + z * sA3;
  const float* Bb = Bm + bb * sB1 + hh * sB2 + z * sB3;
  float*       Cb = C  + bb * sC1 + hh * sC2 + z * sC3;

  const int half = lane >> 4;                // A: K-half this lane holds
  const int r    = lane & 15;
  int arow = m0 + r; if (arow > M - 1) arow = M - 1;   // clamp (M may be 8)
  const float* Arow = Ab + arow * lda;

  v8f acc0 = {}, acc1 = {}, acc2 = {}, acc3 = {};

  for (int k0 = 0; k0 < Kd; k0 += 32) {
    // ---- A fragment: 16x32 (MxK). lane<16 holds K {k0..k0+15},
    //      lane>=16 holds K {k0+16..k0+31} (matches B).
    v16h a;
    if (GUARD) {
      const int ka = k0 + half * 16;
      #pragma unroll
      for (int i = 0; i < 8; ++i) {
        const int kk  = ka + i;
        const int kk2 = kk + 8;
        const float v0 = Arow[kk  < Kd ? kk  : 0];   // always-valid address
        const float v1 = Arow[kk2 < Kd ? kk2 : 0];
        a[i]     = (_Float16)(kk  < Kd ? v0 : 0.f);  // value select, no branch
        a[8 + i] = (_Float16)(kk2 < Kd ? v1 : 0.f);
      }
    } else {
      const float* p0 = Arow + k0 + half * 16;
      if (k0 + 32 < Kd) __builtin_prefetch(p0 + 32, 0, 0);
      f32x4 u0 = *(const f32x4*)(p0);
      f32x4 u1 = *(const f32x4*)(p0 + 4);
      f32x4 u2 = *(const f32x4*)(p0 + 8);
      f32x4 u3 = *(const f32x4*)(p0 + 12);
      #pragma unroll
      for (int i = 0; i < 4; ++i) {
        a[i]      = (_Float16)u0[i];
        a[4 + i]  = (_Float16)u1[i];
        a[8 + i]  = (_Float16)u2[i];
        a[12 + i] = (_Float16)u3[i];
      }
    }

    // ---- B fragment loader: 32x16 (KxN). lane holds column (lane&15),
    //      K = k0 + (lane>>4)*16 + j.
    auto loadB = [&](int t) -> v16h {
      v16h bf;
      int col = n0 + t * 16 + r;
      if (GUARD && col > N - 1) col = N - 1;
      if (TRANSB) {                          // Bm is [N,K] row-major (x @ W^T)
        const float* p = Bb + col * ldb + k0 + half * 16;
        if (GUARD) {
          #pragma unroll
          for (int i = 0; i < 16; ++i) {
            const int kk = k0 + half * 16 + i;
            const float v = Bb[col * ldb + (kk < Kd ? kk : 0)];
            bf[i] = (_Float16)(kk < Kd ? v : 0.f);
          }
        } else {
          f32x4 w0 = *(const f32x4*)(p);
          f32x4 w1 = *(const f32x4*)(p + 4);
          f32x4 w2 = *(const f32x4*)(p + 8);
          f32x4 w3 = *(const f32x4*)(p + 12);
          #pragma unroll
          for (int i = 0; i < 4; ++i) {
            bf[i]      = (_Float16)w0[i];
            bf[4 + i]  = (_Float16)w1[i];
            bf[8 + i]  = (_Float16)w2[i];
            bf[12 + i] = (_Float16)w3[i];
          }
        }
      } else {                               // Bm is [K,N] row-major (P @ V)
        #pragma unroll
        for (int i = 0; i < 16; ++i) {
          const int kk = k0 + half * 16 + i;
          const int kc = GUARD ? (kk < Kd ? kk : 0) : kk;
          const float v = Bb[kc * ldb + col];
          bf[i] = (_Float16)((!GUARD || kk < Kd) ? v : 0.f);
        }
      }
      return bf;
    };

    acc0 = __builtin_amdgcn_wmma_f32_16x16x32_f16(false, a, false, loadB(0),
                                                  (short)0, acc0, false, false);
    acc1 = __builtin_amdgcn_wmma_f32_16x16x32_f16(false, a, false, loadB(1),
                                                  (short)0, acc1, false, false);
    acc2 = __builtin_amdgcn_wmma_f32_16x16x32_f16(false, a, false, loadB(2),
                                                  (short)0, acc2, false, false);
    acc3 = __builtin_amdgcn_wmma_f32_16x16x32_f16(false, a, false, loadB(3),
                                                  (short)0, acc3, false, false);
  }

  // ---- epilogue. C layout: VGPR j -> row m0 + (lane>>4)*8 + j, col lane&15.
  const int rb = bb * sR1 + z * sR3;
  auto store = [&](const v8f& acc, int t) {
    const int col = n0 + t * 16 + r;
    if (col >= N) return;
    const float bv = bias ? bias[col] : 0.f;
    #pragma unroll
    for (int j = 0; j < 8; ++j) {
      const int row = m0 + half * 8 + j;
      if (row >= M) continue;
      float v = acc[j] * alpha + bv;
      if (RELU) v = fmaxf(v, 0.f);
      if (Rp) v += Rp[rb + row * ldr + col];
      Cb[row * ldc + col] = v;
    }
  };
  store(acc0, 0); store(acc1, 1); store(acc2, 2); store(acc3, 3);
}

// ---------------------------------------------------------------------------
// Row softmax (wave per row, in place).
// ---------------------------------------------------------------------------
__global__ __launch_bounds__(256) void softmax_k(float* __restrict__ X,
                                                 int len, int nrows) {
  const int row = blockIdx.x * 8 + (threadIdx.x >> 5);
  if (row >= nrows) return;
  const int lane = threadIdx.x & 31;
  float* p = X + (long)row * len;
  float mx = -3.0e38f;
  for (int i = lane; i < len; i += 32) mx = fmaxf(mx, p[i]);
  #pragma unroll
  for (int o = 16; o > 0; o >>= 1) mx = fmaxf(mx, __shfl_xor(mx, o, 32));
  float s = 0.f;
  for (int i = lane; i < len; i += 32) { float e = __expf(p[i] - mx); p[i] = e; s += e; }
  #pragma unroll
  for (int o = 16; o > 0; o >>= 1) s += __shfl_xor(s, o, 32);
  const float inv = 1.f / s;
  for (int i = lane; i < len; i += 32) p[i] *= inv;
}

// ---------------------------------------------------------------------------
// y = LayerNorm(x + r) * g + b   (row length D=512, wave per row; y may = x)
// ---------------------------------------------------------------------------
__global__ __launch_bounds__(256) void add_ln_k(
    const float* __restrict__ x, const float* __restrict__ rr,
    const float* __restrict__ g, const float* __restrict__ bta,
    float* __restrict__ y, int nrows) {
  const int row = blockIdx.x * 8 + (threadIdx.x >> 5);
  if (row >= nrows) return;
  const int lane = threadIdx.x & 31;
  const float* px = x  + (long)row * D;
  const float* pr = rr + (long)row * D;
  float v[16];
  float s = 0.f;
  #pragma unroll
  for (int i = 0; i < 16; ++i) { v[i] = px[lane + 32*i] + pr[lane + 32*i]; s += v[i]; }
  #pragma unroll
  for (int o = 16; o > 0; o >>= 1) s += __shfl_xor(s, o, 32);
  const float mean = s * (1.f / (float)D);
  float q = 0.f;
  #pragma unroll
  for (int i = 0; i < 16; ++i) { float d = v[i] - mean; q += d * d; }
  #pragma unroll
  for (int o = 16; o > 0; o >>= 1) q += __shfl_xor(q, o, 32);
  const float rstd = rsqrtf(q * (1.f / (float)D) + 1e-5f);
  float* py = y + (long)row * D;
  #pragma unroll
  for (int i = 0; i < 16; ++i) {
    const int col = lane + 32*i;
    py[col] = (v[i] - mean) * rstd * g[col] + bta[col];
  }
}

// ---------------------------------------------------------------------------
// seqs = item_emb[log_seqs] + pos_emb[l]
// ---------------------------------------------------------------------------
__global__ void embed_k(const int* __restrict__ logs,
                        const float* __restrict__ item,
                        const float* __restrict__ pe,
                        float* __restrict__ y) {
  long i = (long)blockIdx.x * blockDim.x + threadIdx.x;
  if (i >= (long)NB * SEQ * D) return;
  int d = (int)(i % D);
  long t = i / D;
  int l = (int)(t % SEQ);
  int id = logs[t];
  y[i] = item[(long)id * D + d] + pe[(long)l * D + d];
}

// ---------------------------------------------------------------------------
// avg_pool1d(kernel=8, stride=8) over the sequence dim.
// ---------------------------------------------------------------------------
__global__ void avgpool8_k(const float* __restrict__ x, float* __restrict__ y,
                           int Ss, long total) {
  long i = (long)blockIdx.x * blockDim.x + threadIdx.x;
  if (i >= total) return;
  int d = (int)(i % D);
  long t = i / D;
  int s = (int)(t % Ss);
  int b = (int)(t / Ss);
  const float* p = x + ((long)(b * Ss + s) * 8) * D + d;
  float acc = 0.f;
  #pragma unroll
  for (int j = 0; j < 8; ++j) acc += p[(long)j * D];
  y[i] = acc * 0.125f;
}

// ---------------------------------------------------------------------------
// pos/neg logits: dot(feats[b,l,:], item_emb[idx]) ; wave per (b,l)
// ---------------------------------------------------------------------------
__global__ __launch_bounds__(256) void logits_k(
    const float* __restrict__ feats, const float* __restrict__ item,
    const int* __restrict__ pos, const int* __restrict__ neg,
    float* __restrict__ out) {
  const int row = blockIdx.x * 8 + (threadIdx.x >> 5);
  if (row >= NB * SEQ) return;
  const int lane = threadIdx.x & 31;
  const float* f  = feats + (long)row * D;
  const float* pi = item + (long)pos[row] * D;
  const float* ni = item + (long)neg[row] * D;
  float sp = 0.f, sn = 0.f;
  for (int i = lane; i < D; i += 32) {
    float fv = f[i];
    sp += fv * pi[i];
    sn += fv * ni[i];
  }
  #pragma unroll
  for (int o = 16; o > 0; o >>= 1) {
    sp += __shfl_xor(sp, o, 32);
    sn += __shfl_xor(sn, o, 32);
  }
  if (lane == 0) { out[row] = sp; out[(long)NB * SEQ + row] = sn; }
}

// ===========================================================================
// Host-side orchestration
// ===========================================================================
struct GP {
  const float *A = nullptr;  int lda = 0, sA1 = 0, sA2 = 0, sA3 = 0;
  const float *Bm = nullptr; int ldb = 0, sB1 = 0, sB2 = 0, sB3 = 0;
  const float *bias = nullptr;
  const float *Rp = nullptr; int ldr = 0, sR1 = 0, sR3 = 0;
  float *C = nullptr;        int ldc = 0, sC1 = 0, sC2 = 0, sC3 = 0;
  int M = 0, N = 0, Kd = 0;  float alpha = 1.f;
  int g0 = 0, HDIV = 1, batches = 1;
};

static inline int imin(int a, int b) { return a < b ? a : b; }

static void run_gemm(hipStream_t st, const GP& p, bool transb, bool relu, bool guard) {
  dim3 bl(256);
  dim3 gr((unsigned)((p.N + 63) / 64), (unsigned)((p.M + 127) / 128),
          (unsigned)p.batches);
#define CALLG(T, R, G)                                                        \
  gemm_k<T, R, G><<<gr, bl, 0, st>>>(                                          \
      p.A, p.lda, p.sA1, p.sA2, p.sA3, p.Bm, p.ldb, p.sB1, p.sB2, p.sB3,       \
      p.bias, p.Rp, p.ldr, p.sR1, p.sR3, p.C, p.ldc, p.sC1, p.sC2, p.sC3,      \
      p.M, p.N, p.Kd, p.alpha, p.g0, p.HDIV)
  if (transb) {
    if (relu)  { if (guard) CALLG(true, true,  true); else CALLG(true, true,  false); }
    else       { if (guard) CALLG(true, false, true); else CALLG(true, false, false); }
  } else       { if (guard) CALLG(false, false, true); else CALLG(false, false, false); }
#undef CALLG
}

struct AttnW { const float *qkv_w, *qkv_b, *out_w, *out_b; };
struct LnW   { const float *g, *b; };
struct FfnW  { const float *w1, *b1, *w2, *b2; };

struct Ctx {
  hipStream_t st;
  float *X0, *P0, *T0, *T1, *Qb, *Kb, *Vb, *Ob, *SC;
  float *XS1, *X1E, *XP1, *XS2, *X2E, *XP2;
  AttnW encA, decSelf, decCross;
  LnW encLn1, encLn2, decLn1, decLn2, decLn3;
  FfnW encF, decF;
  const float *up_w, *up_b;
};

static void linear(hipStream_t st, const float* x, int Mrows, int Kd,
                   const float* W, int N, const float* bias, float* y,
                   bool relu) {
  GP p;
  p.A = x; p.lda = Kd;
  p.Bm = W; p.ldb = Kd;
  p.bias = bias;
  p.C = y; p.ldc = N;
  p.M = Mrows; p.N = N; p.Kd = Kd;
  run_gemm(st, p, true, relu, false);
}

static void softmax_rows(hipStream_t st, float* X, long nrows, int len) {
  unsigned blocks = (unsigned)((nrows + 7) / 8);
  softmax_k<<<dim3(blocks), 256, 0, st>>>(X, len, (int)nrows);
}

static void add_ln(hipStream_t st, const float* x, const float* r,
                   const LnW& w, float* y, int nrows) {
  unsigned blocks = (unsigned)((nrows + 7) / 8);
  add_ln_k<<<dim3(blocks), 256, 0, st>>>(x, r, w.g, w.b, y, nrows);
}

static void ffn(Ctx& c, const float* x, float* y, int Ntok, const FfnW& w) {
  const int CH = 8192;                           // hidden chunk fits in SC
  for (int r0 = 0; r0 < Ntok; r0 += CH) {
    int rows = imin(CH, Ntok - r0);
    linear(c.st, x + (long)r0 * D, rows, D, w.w1, DFF, w.b1, c.SC, true);
    linear(c.st, c.SC, rows, DFF, w.w2, D, w.b2, y + (long)r0 * D, false);
  }
}

static void mha(Ctx& c, const float* qin, int Sq, const float* kvin, int Sk,
                const AttnW& w, float* out) {
  linear(c.st, qin,  NB * Sq, D, w.qkv_w,                D, w.qkv_b,         c.Qb, false);
  linear(c.st, kvin, NB * Sk, D, w.qkv_w + (long)D * D,  D, w.qkv_b + D,     c.Kb, false);
  linear(c.st, kvin, NB * Sk, D, w.qkv_w + 2L * D * D,   D, w.qkv_b + 2 * D, c.Vb, false);
  const int G = NB * NHEAD;
  const int scE = Sq * Sk;
  int maxG = (int)(SC_FLOATS / scE); if (maxG > G) maxG = G;
  const bool gN = (Sk % 16) != 0, gK = (Sk % 32) != 0;
  for (int g0 = 0; g0 < G; g0 += maxG) {
    int gc = imin(maxG, G - g0);
    { // scores = (1/8) * q @ k^T, batched over (b,h)
      GP p;
      p.A = c.Qb; p.lda = D; p.sA1 = Sq * D; p.sA2 = HDIM;
      p.Bm = c.Kb; p.ldb = D; p.sB1 = Sk * D; p.sB2 = HDIM;
      p.C = c.SC; p.ldc = Sk; p.sC3 = scE;
      p.M = Sq; p.N = Sk; p.Kd = HDIM; p.alpha = 0.125f;
      p.g0 = g0; p.HDIV = NHEAD; p.batches = gc;
      run_gemm(c.st, p, true, false, gN);
    }
    softmax_rows(c.st, c.SC, (long)gc * Sq, Sk);
    { // O[...,h] = P @ V
      GP p;
      p.A = c.SC; p.lda = Sk; p.sA3 = scE;
      p.Bm = c.Vb; p.ldb = D; p.sB1 = Sk * D; p.sB2 = HDIM;
      p.C = c.Ob; p.ldc = D; p.sC1 = Sq * D; p.sC2 = HDIM;
      p.M = Sq; p.N = HDIM; p.Kd = Sk; p.alpha = 1.f;
      p.g0 = g0; p.HDIV = NHEAD; p.batches = gc;
      run_gemm(c.st, p, false, false, gK);
    }
  }
  linear(c.st, c.Ob, NB * Sq, D, w.out_w, D, w.out_b, out, false);
}

static void encoder(Ctx& c, const float* xin, float* xout, int S) {
  mha(c, xin, S, xin, S, c.encA, c.T0);
  add_ln(c.st, xin, c.T0, c.encLn1, c.T1, NB * S);
  ffn(c, c.T1, c.T0, NB * S, c.encF);
  add_ln(c.st, c.T1, c.T0, c.encLn2, xout, NB * S);
}

static void decoder(Ctx& c, const float* tgt, const float* mem, int St, int Sm,
                    float* xout) {
  mha(c, tgt, St, tgt, St, c.decSelf, c.T0);
  add_ln(c.st, tgt, c.T0, c.decLn1, c.T1, NB * St);
  mha(c, c.T1, St, mem, Sm, c.decCross, c.T0);
  add_ln(c.st, c.T1, c.T0, c.decLn2, c.T1, NB * St);   // in-place safe
  ffn(c, c.T1, c.T0, NB * St, c.decF);
  add_ln(c.st, c.T1, c.T0, c.decLn3, xout, NB * St);
}

static void down(Ctx& c, const float* x, int S, float* xs) {
  const int Ss = S / 8;
  long total = (long)NB * Ss * D;
  avgpool8_k<<<dim3((unsigned)((total + 255) / 256)), 256, 0, c.st>>>(x, xs, Ss, total);
  { // scores[b] = sh @ x^T
    GP p;
    p.A = xs; p.lda = D; p.sA1 = Ss * D;
    p.Bm = x; p.ldb = D; p.sB1 = S * D;
    p.C = c.SC; p.ldc = S; p.sC1 = Ss * S;
    p.M = Ss; p.N = S; p.Kd = D; p.batches = NB;
    run_gemm(c.st, p, true, false, false);
  }
  softmax_rows(c.st, c.SC, (long)NB * Ss, S);
  { // xs = xs + w @ x     (in place: each element read-once then written)
    GP p;
    p.A = c.SC; p.lda = S; p.sA1 = Ss * S;
    p.Bm = x; p.ldb = D; p.sB1 = S * D;
    p.Rp = xs; p.ldr = D; p.sR1 = Ss * D;
    p.C = xs; p.ldc = D; p.sC1 = Ss * D;
    p.M = Ss; p.N = D; p.Kd = S; p.batches = NB;
    run_gemm(c.st, p, false, false, false);
  }
}

static void up(Ctx& c, const float* xp, const float* xo, int S, float* xu) {
  const int Ss = S / 8;
  linear(c.st, xp, NB * Ss, D, c.up_w, 8 * D, c.up_b, xu, false);  // -> [B,S,D]
  { // scores[b] = up @ xo^T
    GP p;
    p.A = xu; p.lda = D; p.sA1 = S * D;
    p.Bm = xo; p.ldb = D; p.sB1 = S * D;
    p.C = c.SC; p.ldc = S; p.sC1 = S * S;
    p.M = S; p.N = S; p.Kd = D; p.batches = NB;
    run_gemm(c.st, p, true, false, false);
  }
  softmax_rows(c.st, c.SC, (long)NB * S, S);
  { // xu = xu + w @ xo
    GP p;
    p.A = c.SC; p.lda = S; p.sA1 = S * S;
    p.Bm = xo; p.ldb = D; p.sB1 = S * D;
    p.Rp = xu; p.ldr = D; p.sR1 = S * D;
    p.C = xu; p.ldc = D; p.sC1 = S * D;
    p.M = S; p.N = D; p.Kd = S; p.batches = NB;
    run_gemm(c.st, p, false, false, false);
  }
}

extern "C" void kernel_launch(void* const* d_in, const int* in_sizes, int n_in,
                              void* d_out, int out_size, void* d_ws, size_t ws_size,
                              hipStream_t stream) {
  (void)in_sizes; (void)n_in; (void)out_size; (void)ws_size;
  const int* logs = (const int*)d_in[1];
  const int* poss = (const int*)d_in[2];
  const int* negs = (const int*)d_in[3];
  const float* item = (const float*)d_in[4];
  const float* pemb = (const float*)d_in[5];
  auto F = [&](int i) { return (const float*)d_in[i]; };

  Ctx c; c.st = stream;
  float* wbase = (float*)d_ws;
  long off = 0;
  auto alloc = [&](long n) { float* p = wbase + off; off += n; return p; };
  const long SZ0 = (long)NB * SEQ * D;   // 16.8M f32
  const long SZ1 = (long)NB * 64 * D;    // 2.1M
  const long SZ2 = (long)NB * 8 * D;     // 0.26M
  c.X0 = alloc(SZ0); c.P0 = alloc(SZ0); c.T0 = alloc(SZ0); c.T1 = alloc(SZ0);
  c.Qb = alloc(SZ0); c.Kb = alloc(SZ0); c.Vb = alloc(SZ0); c.Ob = alloc(SZ0);
  c.SC = alloc(SC_FLOATS);
  c.XS1 = alloc(SZ1); c.X1E = alloc(SZ1); c.XP1 = alloc(SZ1);
  c.XS2 = alloc(SZ2); c.X2E = alloc(SZ2); c.XP2 = alloc(SZ2);

  c.encA     = { F(6),  F(7),  F(8),  F(9)  };
  c.encF     = { F(10), F(11), F(12), F(13) };
  c.encLn1   = { F(14), F(15) };
  c.encLn2   = { F(16), F(17) };
  c.decSelf  = { F(18), F(19), F(20), F(21) };
  c.decCross = { F(22), F(23), F(24), F(25) };
  c.decF     = { F(26), F(27), F(28), F(29) };
  c.decLn1   = { F(30), F(31) };
  c.decLn2   = { F(32), F(33) };
  c.decLn3   = { F(34), F(35) };
  c.up_w = F(36); c.up_b = F(37);

  { // embedding gather + positional add
    long total = (long)NB * SEQ * D;
    embed_k<<<dim3((unsigned)((total + 255) / 256)), 256, 0, stream>>>(
        logs, item, pemb, c.X0);
  }

  // hourglass with sf = (8, 8)
  encoder(c, c.X0, c.P0, SEQ);            // x0e
  down(c, c.P0, SEQ, c.XS1);              // -> [B,64,D]
  encoder(c, c.XS1, c.X1E, 64);           // x1e
  down(c, c.X1E, 64, c.XS2);              // -> [B,8,D]
  encoder(c, c.XS2, c.X2E, 8);            // x2e
  decoder(c, c.X2E, c.X2E, 8, 8, c.XP2);  // innermost dec
  up(c, c.XP2, c.X1E, 64, c.XS1);         // xu1 (XS1 reused)
  decoder(c, c.XS1, c.X1E, 64, 64, c.XP1);
  up(c, c.XP1, c.P0, SEQ, c.X0);          // xu0 (X0 reused)
  decoder(c, c.X0, c.P0, SEQ, SEQ, c.X0); // feats in X0

  logits_k<<<dim3((unsigned)((NB * SEQ) / 8)), 256, 0, stream>>>(
      c.X0, item, poss, negs, (float*)d_out);
}
